// SwinEncoder_80212809220263
// MI455X (gfx1250) — matrix operand order repaired and checked by reading, not compiled, not run on hardware
//
#include <hip/hip_runtime.h>
#include <hip/hip_bf16.h>
#include <math.h>

typedef __attribute__((ext_vector_type(16))) _Float16 v16h;
typedef __attribute__((ext_vector_type(8)))  float    v8f;

// ---------------------------------------------------------------------------
// f32 -> f16 conversion
// ---------------------------------------------------------------------------
__global__ __launch_bounds__(256) void k_f32_to_f16(const float* __restrict__ in,
                                                    _Float16* __restrict__ out, int n) {
    int i = blockIdx.x * 256 + threadIdx.x;
    if (i < n) out[i] = (_Float16)in[i];
}

__global__ __launch_bounds__(256) void k_copy(const float* __restrict__ in,
                                              float* __restrict__ out, size_t n) {
    size_t i = (size_t)blockIdx.x * 256 + threadIdx.x;
    if (i < n) out[i] = in[i];
}

// ---------------------------------------------------------------------------
// Patch embedding: 4x4/4 conv, 3->128, NCHW input -> (B,56,56,128) token-major
// ---------------------------------------------------------------------------
__global__ __launch_bounds__(256) void k_patch_embed(const float* __restrict__ x,
                                                     const float* __restrict__ w,
                                                     float* __restrict__ out) {
    size_t idx = (size_t)blockIdx.x * 256 + threadIdx.x;
    int oc = idx % 128; size_t t = idx / 128;
    int ow = t % 56; size_t t2 = t / 56;
    int oh = t2 % 56; int b = (int)(t2 / 56);
    float acc = 0.f;
    for (int ic = 0; ic < 3; ic++)
        for (int kh = 0; kh < 4; kh++)
            for (int kw = 0; kw < 4; kw++)
                acc += x[(((size_t)b * 3 + ic) * 224 + (oh * 4 + kh)) * 224 + (ow * 4 + kw)] *
                       w[((oc * 3 + ic) * 4 + kh) * 4 + kw];
    out[idx] = acc;
}

// ---------------------------------------------------------------------------
// LayerNorm: one wave per token, f32 in -> f16 out
// ---------------------------------------------------------------------------
__global__ __launch_bounds__(256) void k_layernorm(const float* __restrict__ x,
                                                   const float* __restrict__ g,
                                                   const float* __restrict__ b,
                                                   _Float16* __restrict__ out, int T, int C) {
    int lane = threadIdx.x & 31, wave = threadIdx.x >> 5;
    int tok = blockIdx.x * 8 + wave;
    if (tok >= T) return;
    const float* xr = x + (size_t)tok * C;
    float s = 0.f, s2 = 0.f;
    for (int c = lane; c < C; c += 32) { float v = xr[c]; s += v; s2 += v * v; }
    for (int off = 16; off > 0; off >>= 1) {
        s  += __shfl_xor(s,  off, 32);
        s2 += __shfl_xor(s2, off, 32);
    }
    float mu = s / C;
    float rstd = rsqrtf(fmaxf(s2 / C - mu * mu, 0.f) + 1e-5f);
    _Float16* orow = out + (size_t)tok * C;
    for (int c = lane; c < C; c += 32)
        orow[c] = (_Float16)((xr[c] - mu) * rstd * g[c] + b[c]);
}

// ---------------------------------------------------------------------------
// WMMA GEMM: Y[M,N] = A[M,K] * W[N,K]^T (+bias)(+GELU)(+residual)
// A,W f16 row-major. Per-WG 64x64 tile, 8 waves, each wave 16x32 via two
// v_wmma_f32_16x16x32_f16. Fragment layouts per CDNA5 ISA 7.12.2.
// ---------------------------------------------------------------------------
__global__ __launch_bounds__(256) void k_gemm(const _Float16* __restrict__ A,
                                              const _Float16* __restrict__ W,
                                              const float* __restrict__ bias,
                                              const float* __restrict__ resid,
                                              float* __restrict__ outF,
                                              _Float16* __restrict__ outH,
                                              int M, int N, int K, int act) {
    int lane = threadIdx.x & 31, wave = threadIdx.x >> 5;
    int lrow = lane & 15, hi = lane >> 4;
    int row0 = blockIdx.y * 64 + (wave & 3) * 16;
    int col0 = blockIdx.x * 64 + (wave >> 2) * 32;
    const _Float16* ar = A + (size_t)(row0 + lrow) * K;
    const _Float16* w0 = W + (size_t)(col0 + lrow) * K;
    const _Float16* w1 = W + (size_t)(col0 + 16 + lrow) * K;
    int kbA = hi * 8;    // lanes 16-31 hold K+8..K+15 halves of A rows
    int kbB = hi * 8;    // lanes 16-31 hold K+8..K+15 halves of B columns
    v8f c0 = {}, c1 = {};
    for (int k = 0; k < K; k += 32) {
        __builtin_prefetch(ar + k + 128, 0, 1);   // global_prefetch_b8
        v16h a, b0, b1;
        *(float4*)&a         = *(const float4*)(ar + k + kbA);
        *(((float4*)&a) + 1) = *(const float4*)(ar + k + 16 + kbA);
        *(float4*)&b0        = *(const float4*)(w0 + k + kbB);
        *(((float4*)&b0) + 1) = *(const float4*)(w0 + k + kbB + 16);
        *(float4*)&b1        = *(const float4*)(w1 + k + kbB);
        *(((float4*)&b1) + 1) = *(const float4*)(w1 + k + kbB + 16);
        c0 = __builtin_amdgcn_wmma_f32_16x16x32_f16(false, a, false, b0, (short)0, c0, false, false);
        c1 = __builtin_amdgcn_wmma_f32_16x16x32_f16(false, a, false, b1, (short)0, c1, false, false);
    }
#pragma unroll
    for (int i = 0; i < 8; i++) {
        int r = row0 + hi * 8 + i;
#pragma unroll
        for (int t = 0; t < 2; t++) {
            int cc = col0 + t * 16 + lrow;
            float v = t ? c1[i] : c0[i];
            if (bias)  v += bias[cc];
            if (act)   v = 0.5f * v * (1.f + erff(v * 0.70710678118654752f));   // exact GELU
            if (resid) v += resid[(size_t)r * N + cc];
            size_t oi = (size_t)r * N + cc;
            if (outF) outF[oi] = v;
            if (outH) outH[oi] = (_Float16)v;
        }
    }
}

// ---------------------------------------------------------------------------
// Window partition (with -shift roll folded in): f16 LN output -> (nWin,64,C),
// rows 49..63 zero-padded.
// ---------------------------------------------------------------------------
__global__ __launch_bounds__(256) void k_window_part(const _Float16* __restrict__ xln,
                                                     _Float16* __restrict__ win,
                                                     int Himg, int C, int shift, int nW) {
    size_t idx = (size_t)blockIdx.x * 256 + threadIdx.x;
    int c = idx % C; size_t t = idx / C;
    int n = t % 64; size_t wg = t / 64;
    int wx = wg % nW; size_t wg2 = wg / nW;
    int wy = wg2 % nW; int b = (int)(wg2 / nW);
    _Float16 v = (_Float16)0.f;
    if (n < 49) {
        int iy = n / 7, ix = n % 7;
        int h = (wy * 7 + iy + shift) % Himg;
        int w = (wx * 7 + ix + shift) % Himg;
        v = xln[(((size_t)b * Himg + h) * Himg + w) * C + c];
    }
    win[idx] = v;
}

// ---------------------------------------------------------------------------
// Un-window (+shift roll back) + residual add into x (in place)
// ---------------------------------------------------------------------------
__global__ __launch_bounds__(256) void k_unwindow_add(float* __restrict__ x,
                                                      const float* __restrict__ po,
                                                      int Himg, int C, int shift, int nW) {
    size_t idx = (size_t)blockIdx.x * 256 + threadIdx.x;
    int c = idx % C; size_t t = idx / C;
    int w = t % Himg; size_t t2 = t / Himg;
    int h = t2 % Himg; int b = (int)(t2 / Himg);
    int hr = (h - shift + Himg) % Himg;
    int wr = (w - shift + Himg) % Himg;
    size_t winI = ((size_t)b * nW + hr / 7) * nW + wr / 7;
    int n = (hr % 7) * 7 + (wr % 7);
    x[idx] += po[(winI * 64 + n) * C + c];
}

// ---------------------------------------------------------------------------
// Window attention for one (window, head): S = (q kT)/sqrt(32) + bias + mask,
// softmax over 49 valid cols, out = P V. hd == 32 in all stages.
// q/k staged row-major in LDS, V staged TRANSPOSED in LDS so every WMMA
// fragment is two contiguous 16B LDS reads per lane.
// ---------------------------------------------------------------------------
__global__ __launch_bounds__(256) void k_attn(const _Float16* __restrict__ qkv,
                                              const float* __restrict__ rel_bias,
                                              _Float16* __restrict__ out,
                                              int C, int heads, int nW, int shift, int Himg) {
    __shared__ float    S[64][65];
    __shared__ _Float16 P[64][64];
    __shared__ _Float16 Qs[64][32];
    __shared__ _Float16 Ks[64][32];
    __shared__ _Float16 Vt[32][64];
    int head = blockIdx.x % heads;
    int win  = blockIdx.x / heads;
    int lane = threadIdx.x & 31, wave = threadIdx.x >> 5;
    int lrow = lane & 15, hi = lane >> 4;
    int C3 = 3 * C;
    const _Float16* qb = qkv + (size_t)win * 64 * C3 + head * 32;
    const _Float16* kb = qb + C;
    const _Float16* vb = qb + 2 * C;

    // ---- stage q, k, v(T) into LDS: one b128 per thread per matrix ----
    {
        int t8 = threadIdx.x * 8;          // 256 threads * 8 halves = 64x32
        int r  = t8 >> 5;                  // row 0..63
        int c0 = t8 & 31;                  // col 0,8,16,24
        *(float4*)&Qs[r][c0] = *(const float4*)(qb + (size_t)r * C3 + c0);
        *(float4*)&Ks[r][c0] = *(const float4*)(kb + (size_t)r * C3 + c0);
        float4 vv = *(const float4*)(vb + (size_t)r * C3 + c0);
        const _Float16* vh = (const _Float16*)&vv;
#pragma unroll
        for (int e = 0; e < 8; e++) Vt[c0 + e][r] = vh[e];
    }
    __syncthreads();

    // ---- phase 1: scores, 16 tiles of 16x16, K=32, two tiles per wave ----
    for (int t = 2 * wave; t <= 2 * wave + 1; t++) {
        int ti = t >> 2, tj = t & 3;
        v16h a, bf;
        *(float4*)&a          = *(const float4*)&Qs[ti * 16 + lrow][hi * 8];
        *(((float4*)&a) + 1)  = *(const float4*)&Qs[ti * 16 + lrow][16 + hi * 8];
        *(float4*)&bf         = *(const float4*)&Ks[tj * 16 + lrow][hi * 8];
        *(((float4*)&bf) + 1) = *(const float4*)&Ks[tj * 16 + lrow][16 + hi * 8];
        v8f c = {};
        c = __builtin_amdgcn_wmma_f32_16x16x32_f16(false, a, false, bf, (short)0, c, false, false);
#pragma unroll
        for (int i = 0; i < 8; i++)
            S[ti * 16 + hi * 8 + i][tj * 16 + lrow] = c[i] * 0.1767766952966369f; // 1/sqrt(32)
    }
    __syncthreads();

    // ---- phase 2: bias + shift mask + softmax (rows 0..48) ----
    if (threadIdx.x < 64) {
        int r = threadIdx.x;
        if (r < 49) {
            int wg = win % (nW * nW);
            int wy = wg / nW, wx = wg % nW;
            int iy = r / 7, ix = r % 7;
            int reg_r = 0;
            if (shift > 0) {
                int hr = wy * 7 + iy, wr = wx * 7 + ix;
                reg_r = ((hr < Himg - 7) ? 0 : ((hr < Himg - 3) ? 1 : 2)) * 3 +
                        ((wr < Himg - 7) ? 0 : ((wr < Himg - 3) ? 1 : 2));
            }
            float mx = -1e30f;
            for (int m = 0; m < 49; m++) {
                int jy = m / 7, jx = m % 7;
                float v = S[r][m] +
                          rel_bias[((iy - jy + 6) * 13 + (ix - jx + 6)) * heads + head];
                if (shift > 0) {
                    int hm = wy * 7 + jy, wm = wx * 7 + jx;
                    int reg_m = ((hm < Himg - 7) ? 0 : ((hm < Himg - 3) ? 1 : 2)) * 3 +
                                ((wm < Himg - 7) ? 0 : ((wm < Himg - 3) ? 1 : 2));
                    if (reg_m != reg_r) v -= 100.f;
                }
                S[r][m] = v;
                mx = fmaxf(mx, v);
            }
            float sum = 0.f;
            for (int m = 0; m < 49; m++) { float e = __expf(S[r][m] - mx); S[r][m] = e; sum += e; }
            float inv = 1.f / sum;
            for (int m = 0; m < 49; m++) P[r][m] = (_Float16)(S[r][m] * inv);
            for (int m = 49; m < 64; m++) P[r][m] = (_Float16)0.f;
        } else {
            for (int m = 0; m < 64; m++) P[threadIdx.x][m] = (_Float16)0.f;
        }
    }
    __syncthreads();

    // ---- phase 3: out(64x32) = P(64x64) @ V(64x32), one 16x16 tile / wave ----
    int mt = wave >> 1, nt = wave & 1;
    v8f c = {};
#pragma unroll
    for (int k0 = 0; k0 < 64; k0 += 32) {
        v16h a, bf;
        *(float4*)&a          = *(const float4*)&P[mt * 16 + lrow][k0 + hi * 8];
        *(((float4*)&a) + 1)  = *(const float4*)&P[mt * 16 + lrow][k0 + 16 + hi * 8];
        *(float4*)&bf         = *(const float4*)&Vt[nt * 16 + lrow][k0 + hi * 8];
        *(((float4*)&bf) + 1) = *(const float4*)&Vt[nt * 16 + lrow][k0 + 16 + hi * 8];
        c = __builtin_amdgcn_wmma_f32_16x16x32_f16(false, a, false, bf, (short)0, c, false, false);
    }
#pragma unroll
    for (int i = 0; i < 8; i++) {
        int r = mt * 16 + hi * 8 + i;
        out[((size_t)win * 64 + r) * C + head * 32 + nt * 16 + lrow] = (_Float16)c[i];
    }
}

// ---------------------------------------------------------------------------
// Patch merge gather: (B,H,W,C) -> (B*H/2*W/2, 4C) with [x0,x1,x2,x3] concat
// ---------------------------------------------------------------------------
__global__ __launch_bounds__(256) void k_merge_gather(const float* __restrict__ x,
                                                      float* __restrict__ y, int Hp, int Cp) {
    size_t idx = (size_t)blockIdx.x * 256 + threadIdx.x;
    int C4 = 4 * Cp, H2 = Hp / 2;
    int c4 = idx % C4; size_t t = idx / C4;
    int w2 = t % H2; size_t t2 = t / H2;
    int h2 = t2 % H2; int b = (int)(t2 / H2);
    int q = c4 / Cp, c = c4 % Cp;
    int dh = (q == 1 || q == 3) ? 1 : 0;   // x0:(0,0) x1:(1,0) x2:(0,1) x3:(1,1)
    int dw = (q >= 2) ? 1 : 0;
    y[idx] = x[(((size_t)b * Hp + (h2 * 2 + dh)) * Hp + (w2 * 2 + dw)) * Cp + c];
}

// ===========================================================================
// Host orchestration
// ===========================================================================
extern "C" void kernel_launch(void* const* d_in, const int* in_sizes, int n_in,
                              void* d_out, int out_size, void* d_ws, size_t ws_size,
                              hipStream_t stream) {
    (void)out_size; (void)ws_size; (void)n_in;
    const int B = 16;
    // ---- input mapping: insertion order (x first) or jax-sorted leaves ----
    bool insertion = (in_sizes[0] == 16 * 3 * 224 * 224);
    auto P = [&](int i) -> const float* { return (const float*)d_in[i]; };
    struct Blk { const float *ln1_g,*ln1_b,*qkv_w,*qkv_b,*proj_w,*proj_b,*rel_bias,
                             *ln2_g,*ln2_b,*fc1_w,*fc1_b,*fc2_w,*fc2_b; };
    Blk blk[8];
    const float *xin, *projw, *mg_g[2], *mg_b[2], *mg_w[2];
    if (insertion) {
        xin = P(0); projw = P(1);
        for (int bi = 0; bi < 8; bi++) {
            int o = 2 + bi * 13;
            blk[bi] = { P(o+0),P(o+1),P(o+2),P(o+3),P(o+4),P(o+5),P(o+6),
                        P(o+7),P(o+8),P(o+9),P(o+10),P(o+11),P(o+12) };
        }
        for (int m = 0; m < 2; m++) { int o = 106 + m * 3; mg_g[m]=P(o); mg_b[m]=P(o+1); mg_w[m]=P(o+2); }
    } else {
        // sorted keys: fc1_b,fc1_w,fc2_b,fc2_w,ln1_b,ln1_g,ln2_b,ln2_g,proj_b,proj_w,qkv_b,qkv_w,rel_bias
        for (int bi = 0; bi < 8; bi++) {
            int o = bi * 13;
            blk[bi] = { P(o+5),P(o+4),P(o+11),P(o+10),P(o+9),P(o+8),P(o+12),
                        P(o+7),P(o+6),P(o+1),P(o+0),P(o+3),P(o+2) };
        }
        for (int m = 0; m < 2; m++) { int o = 104 + m * 3; mg_b[m]=P(o); mg_g[m]=P(o+1); mg_w[m]=P(o+2); }
        projw = P(110); xin = P(111);
    }

    // ---- workspace carve (256B aligned) ----
    char* wp = (char*)d_ws;
    auto carve = [&](size_t bytes) -> void* {
        void* p = wp; wp += (bytes + 255) & ~(size_t)255; return p;
    };
    float*     xf   = (float*)    carve((size_t)6422528 * 4);   // activations (T*C, largest at stage 0)
    float*     tmpf = (float*)    carve((size_t)8388608 * 4);   // proj-out windows / merge concat
    _Float16*  lnh  = (_Float16*) carve((size_t)6422528 * 2);   // LN output f16
    _Float16*  bigh = (_Float16*) carve((size_t)25690112 * 2);  // qkv f16 / fc1-out f16
    _Float16*  winh = (_Float16*) carve((size_t)8388608 * 2);   // windowed x f16 / attn-out f16
    _Float16*  wh   = (_Float16*) carve((size_t)1048576 * 2);   // weight staging f16

    // ---- patch embed ----
    k_patch_embed<<<(B * 56 * 56 * 128) / 256, 256, 0, stream>>>(xin, projw, xf);

    int nblocks_arr[3] = {2, 2, 4};
    int C = 128, Himg = 56, heads = 4;
    size_t outOff = 0;
    int bi = 0;
    for (int st = 0; st < 3; st++) {
        if (st > 0) {
            int Cp = C, Hp = Himg, H2 = Hp / 2;
            int T2 = B * H2 * H2, C4 = 4 * Cp, Nn = 2 * Cp;
            k_merge_gather<<<(unsigned)((size_t)T2 * C4 / 256), 256, 0, stream>>>(xf, tmpf, Hp, Cp);
            k_layernorm<<<T2 / 8, 256, 0, stream>>>(tmpf, mg_g[st-1], mg_b[st-1], lnh, T2, C4);
            k_f32_to_f16<<<(Nn * C4) / 256, 256, 0, stream>>>(mg_w[st-1], wh, Nn * C4);
            dim3 g(Nn / 64, T2 / 64);
            k_gemm<<<g, 256, 0, stream>>>(lnh, wh, nullptr, nullptr, xf, nullptr, T2, Nn, C4, 0);
            C = 2 * Cp; Himg = H2; heads *= 2;
        }
        int T = B * Himg * Himg;
        int nW = Himg / 7;
        int nWin = B * nW * nW;
        int Mw = nWin * 64;
        for (int j = 0; j < nblocks_arr[st]; j++, bi++) {
            const Blk& bp = blk[bi];
            int s = (j % 2) ? 3 : 0;
            // attention half
            k_layernorm<<<T / 8, 256, 0, stream>>>(xf, bp.ln1_g, bp.ln1_b, lnh, T, C);
            k_window_part<<<(unsigned)((size_t)Mw * C / 256), 256, 0, stream>>>(lnh, winh, Himg, C, s, nW);
            k_f32_to_f16<<<(3 * C * C) / 256, 256, 0, stream>>>(bp.qkv_w, wh, 3 * C * C);
            { dim3 g(3 * C / 64, Mw / 64);
              k_gemm<<<g, 256, 0, stream>>>(winh, wh, bp.qkv_b, nullptr, nullptr, bigh, Mw, 3 * C, C, 0); }
            k_attn<<<nWin * heads, 256, 0, stream>>>(bigh, bp.rel_bias, winh, C, heads, nW, s, Himg);
            k_f32_to_f16<<<(C * C) / 256, 256, 0, stream>>>(bp.proj_w, wh, C * C);
            { dim3 g(C / 64, Mw / 64);
              k_gemm<<<g, 256, 0, stream>>>(winh, wh, bp.proj_b, nullptr, tmpf, nullptr, Mw, C, C, 0); }
            k_unwindow_add<<<(unsigned)((size_t)T * C / 256), 256, 0, stream>>>(xf, tmpf, Himg, C, s, nW);
            // MLP half
            k_layernorm<<<T / 8, 256, 0, stream>>>(xf, bp.ln2_g, bp.ln2_b, lnh, T, C);
            k_f32_to_f16<<<(4 * C * C) / 256, 256, 0, stream>>>(bp.fc1_w, wh, 4 * C * C);
            { dim3 g(4 * C / 64, T / 64);
              k_gemm<<<g, 256, 0, stream>>>(lnh, wh, bp.fc1_b, nullptr, nullptr, bigh, T, 4 * C, C, 1); }
            k_f32_to_f16<<<(4 * C * C) / 256, 256, 0, stream>>>(bp.fc2_w, wh, 4 * C * C);
            { dim3 g(C / 64, T / 64);
              k_gemm<<<g, 256, 0, stream>>>(bigh, wh, bp.fc2_b, xf, xf, nullptr, T, C, 4 * C, 0); }
        }
        k_copy<<<(unsigned)((size_t)T * C / 256), 256, 0, stream>>>(xf, (float*)d_out + outOff, (size_t)T * C);
        outOff += (size_t)T * C;
    }
}
